// AttentionHead_8744553414885
// MI455X (gfx1250) — matrix operand order repaired and checked by reading, not compiled, not run on hardware
//
#include <hip/hip_runtime.h>
#include <hip/hip_fp16.h>

typedef __attribute__((ext_vector_type(16))) _Float16 v16h;
typedef __attribute__((ext_vector_type(8)))  float    v8f;

union FragU { v16h v; unsigned int u[8]; };

// A fragment (16x32 f16), row-major storage: elem(m,k) at base[m*ld + koff + k]
__device__ __attribute__((always_inline)) inline
v16h load_a_frag(const _Float16* base, int ld, int koff, int lane) {
  const int m = lane & 15, g = lane >> 4;
  FragU f;
#pragma unroll
  for (int p = 0; p < 8; ++p) {
    const int k = koff + 2 * p + 8 * g + ((p >= 4) ? 8 : 0);
    f.u[p] = *(const unsigned int*)(base + m * ld + k);
  }
  return f.v;
}

// B fragment (32x16 f16), n-major storage: elem(k,n) at base[n*ld + koff + k]
__device__ __attribute__((always_inline)) inline
v16h load_b_frag(const _Float16* base, int ld, int koff, int lane) {
  const int n = lane & 15, g = lane >> 4;
  FragU f;
#pragma unroll
  for (int p = 0; p < 8; ++p) {
    const int k = koff + 2 * p + 8 * g + ((p >= 4) ? 8 : 0);
    f.u[p] = *(const unsigned int*)(base + n * ld + k);
  }
  return f.v;
}

__device__ __attribute__((always_inline)) inline
v8f wmma_f16(v16h a, v16h b, v8f c) {
  return __builtin_amdgcn_wmma_f32_16x16x32_f16(false, a, false, b, (short)0, c,
                                                false, false);
}

// gfx1250 async global->LDS 128-bit copy (ASYNCcnt tracked, bypasses VGPRs).
// INST_OFFSET is added to both the LDS and the global address (ISA 08 §4.4).
#define ASYNC_LD_B128(ldsoff, voff, sbase, IMM)                               \
  asm volatile("global_load_async_to_lds_b128 %0, %1, %2 offset:" #IMM        \
               :: "v"(ldsoff), "v"(voff), "s"(sbase) : "memory")

#define WAIT_ASYNCCNT(N) asm volatile("s_wait_asynccnt " #N ::: "memory")

__device__ __attribute__((always_inline)) inline
unsigned lds_off32(const void* p) {  // LDS aperture: low 32 bits = LDS offset
  return (unsigned)(unsigned long long)p;
}

// ---------------- Kernel 1: QKV projection, f32 -> f16, WMMA GEMM ----------
// x:[BT,384] f32, W*:[384,64] f32 -> Qo/Ko:[BT,64] f16, Vt:[B][64][2048] f16.
__global__ void __launch_bounds__(256) proj_qkv_kernel(
    const float* __restrict__ x,
    const float* __restrict__ Wk, const float* __restrict__ Wq,
    const float* __restrict__ Wv,
    _Float16* __restrict__ Qo, _Float16* __restrict__ Ko,
    _Float16* __restrict__ Vt) {
  __shared__ _Float16 xs[128 * 32];      // 8 KB  row-major [row][k]
  __shared__ _Float16 wts[3][64 * 32];   // 12 KB n-major  [n][k]
  __shared__ _Float16 vstage[64 * 128];  // 16 KB [h][row_local] for V transpose

  const int tid = threadIdx.x, lane = tid & 31, w = tid >> 5;
  const int row0 = blockIdx.x * 128;

  v8f accq[4], acck[4], accv[4];
#pragma unroll
  for (int nt = 0; nt < 4; ++nt)
#pragma unroll
    for (int r = 0; r < 8; ++r) { accq[nt][r] = 0.f; acck[nt][r] = 0.f; accv[nt][r] = 0.f; }

  for (int kc = 0; kc < 384; kc += 32) {
    __syncthreads();
    // stage x chunk: 16 contiguous elements/thread -> vectorizable b128
    {
      const int base = tid * 16;                    // 4096 halves total
      const int r = base >> 5, k0 = base & 31;
      const float* xp = x + (size_t)(row0 + r) * 384 + kc + k0;
#pragma unroll
      for (int i = 0; i < 16; ++i) xs[base + i] = (_Float16)xp[i];
    }
    // stage W chunks transposed to n-major (coalesced global reads)
#pragma unroll
    for (int idx = tid; idx < 64 * 32; idx += 256) {
      const int k = idx >> 6, n = idx & 63;
      const size_t go = (size_t)(kc + k) * 64 + n;
      wts[0][n * 32 + k] = (_Float16)Wq[go];
      wts[1][n * 32 + k] = (_Float16)Wk[go];
      wts[2][n * 32 + k] = (_Float16)Wv[go];
    }
    __syncthreads();

    const v16h a = load_a_frag(&xs[w * 16 * 32], 32, 0, lane);
#pragma unroll
    for (int nt = 0; nt < 4; ++nt) {
      const v16h bq = load_b_frag(&wts[0][nt * 16 * 32], 32, 0, lane);
      accq[nt] = wmma_f16(a, bq, accq[nt]);
      const v16h bk = load_b_frag(&wts[1][nt * 16 * 32], 32, 0, lane);
      acck[nt] = wmma_f16(a, bk, acck[nt]);
      const v16h bv = load_b_frag(&wts[2][nt * 16 * 32], 32, 0, lane);
      accv[nt] = wmma_f16(a, bv, accv[nt]);
    }
  }

  const int g = lane >> 4, n = lane & 15;
#pragma unroll
  for (int nt = 0; nt < 4; ++nt)
#pragma unroll
    for (int r = 0; r < 8; ++r) {
      const size_t off = (size_t)(row0 + w * 16 + g * 8 + r) * 64 + nt * 16 + n;
      Qo[off] = (_Float16)accq[nt][r];
      Ko[off] = (_Float16)acck[nt][r];
      // V: transpose through LDS for coalesced [B][H][T] output
      vstage[(nt * 16 + n) * 128 + w * 16 + g * 8 + r] = (_Float16)accv[nt][r];
    }
  __syncthreads();
  {
    const int base = tid * 32;                      // 8192 halves total
    const int h = base >> 7, t0 = base & 127;
    const int bloc = row0 >> 11, rloc = row0 & 2047;
    _Float16* vp = Vt + ((size_t)bloc * 64 + h) * 2048 + rloc + t0;
#pragma unroll
    for (int i = 0; i < 32; ++i) vp[i] = vstage[base + i];
  }
}

// ---------------- Kernel 2: flash attention, causal, wave32 WMMA -----------
// Block = 256 thr = 8 waves; wave owns 16 query rows; key tiles of 64;
// K/V tiles double-buffered via async global->LDS copies.
__global__ void __launch_bounds__(256) attn_kernel(
    const _Float16* __restrict__ Q, const _Float16* __restrict__ K,
    const _Float16* __restrict__ Vt, float* __restrict__ out) {
  __shared__ _Float16 Kbuf[2][64 * 64];   // 16 KB row-major [kpos][h]
  __shared__ _Float16 Vbuf[2][64 * 64];   // 16 KB n-major   [h][kpos]
  __shared__ _Float16 Ps[8][16 * 64];     // 16 KB per-wave P tile

  const int tid = threadIdx.x, lane = tid & 31, w = tid >> 5;
  const int b = blockIdx.x >> 4, rb = blockIdx.x & 15;
  const int row0 = rb * 128;
  const int qrow_w = row0 + w * 16;                 // batch-local first row
  const size_t boff = (size_t)b * 2048 * 64;
  const int g = lane >> 4, nl = lane & 15;
  const float scale = 0.125f;                       // 1/sqrt(64)

  // per-thread async copy assignment: 32 contiguous bytes per array per tile
  const int kq = tid >> 2;                          // 0..63
  const int r16 = (tid & 3) * 16;                   // 0,16,32,48
  const unsigned long long kbase = (unsigned long long)(const void*)(K + boff);
  const unsigned long long vbase = (unsigned long long)(const void*)(Vt + boff);
  const unsigned ldsK[2] = { lds_off32(&Kbuf[0][kq * 64 + r16]),
                             lds_off32(&Kbuf[1][kq * 64 + r16]) };
  const unsigned ldsV[2] = { lds_off32(&Vbuf[0][kq * 64 + r16]),
                             lds_off32(&Vbuf[1][kq * 64 + r16]) };

  auto issue_tile = [&](int j) {
    const int kt0 = j * 64, bsel = j & 1;
    const unsigned kvoff = (unsigned)((((kt0 + kq) * 64) + r16) * 2);
    const unsigned vvoff = (unsigned)(((kq * 2048) + kt0 + r16) * 2);
    ASYNC_LD_B128(ldsK[bsel], kvoff, kbase, 0);
    ASYNC_LD_B128(ldsK[bsel], kvoff, kbase, 16);
    ASYNC_LD_B128(ldsV[bsel], vvoff, vbase, 0);
    ASYNC_LD_B128(ldsV[bsel], vvoff, vbase, 16);
  };

  v16h qf0 = load_a_frag(Q + boff + (size_t)qrow_w * 64, 64, 0, lane);
  v16h qf1 = load_a_frag(Q + boff + (size_t)qrow_w * 64, 64, 32, lane);

  v8f O[4];
  float mrow[8], lrow[8];
#pragma unroll
  for (int r = 0; r < 8; ++r) { mrow[r] = -1e30f; lrow[r] = 0.f; }
#pragma unroll
  for (int nt = 0; nt < 4; ++nt)
#pragma unroll
    for (int r = 0; r < 8; ++r) O[nt][r] = 0.f;

  const int jmax = rb * 2 + 2;                      // key tiles up to diagonal
  issue_tile(0);
  for (int j = 0; j < jmax; ++j) {
    const int kt0 = j * 64;
    if (j + 1 < jmax) {                             // prefetch next tile
      issue_tile(j + 1);
      WAIT_ASYNCCNT(4);                             // tile j landed
    } else {
      WAIT_ASYNCCNT(0);
    }
    __syncthreads();                                // all waves' copies done
    const _Float16* Ks  = &Kbuf[j & 1][0];
    const _Float16* Vts = &Vbuf[j & 1][0];

    if (kt0 <= qrow_w + 15) {                       // wave-uniform: EXEC all-1s
      // S = Q K^T  (16 x 64)
      v8f S[4];
#pragma unroll
      for (int nt = 0; nt < 4; ++nt) {
        v8f s;
#pragma unroll
        for (int r = 0; r < 8; ++r) s[r] = 0.f;
        const v16h b0 = load_b_frag(Ks + nt * 16 * 64, 64, 0, lane);
        s = wmma_f16(qf0, b0, s);
        const v16h b1 = load_b_frag(Ks + nt * 16 * 64, 64, 32, lane);
        s = wmma_f16(qf1, b1, s);
        S[nt] = s;
      }
      // scale + causal mask + per-row running max
      float mnew[8];
#pragma unroll
      for (int r = 0; r < 8; ++r) mnew[r] = mrow[r];
#pragma unroll
      for (int nt = 0; nt < 4; ++nt) {
        const int key = kt0 + nt * 16 + nl;
#pragma unroll
        for (int r = 0; r < 8; ++r) {
          const int qr = qrow_w + g * 8 + r;
          float sv = S[nt][r] * scale;
          sv = (key <= qr) ? sv : -1e30f;
          S[nt][r] = sv;
          mnew[r] = fmaxf(mnew[r], sv);
        }
      }
#pragma unroll
      for (int r = 0; r < 8; ++r) {                 // reduce over 16-lane half
        float v = mnew[r];
        v = fmaxf(v, __shfl_xor(v, 1, 32));
        v = fmaxf(v, __shfl_xor(v, 2, 32));
        v = fmaxf(v, __shfl_xor(v, 4, 32));
        v = fmaxf(v, __shfl_xor(v, 8, 32));
        mnew[r] = v;
      }
      float rsum[8];
#pragma unroll
      for (int r = 0; r < 8; ++r) {
        const float alpha = __expf(mrow[r] - mnew[r]);
        lrow[r] *= alpha;
#pragma unroll
        for (int nt = 0; nt < 4; ++nt) O[nt][r] *= alpha;
        mrow[r] = mnew[r];
        rsum[r] = 0.f;
      }
#pragma unroll
      for (int nt = 0; nt < 4; ++nt)
#pragma unroll
        for (int r = 0; r < 8; ++r) {
          const float p = __expf(S[nt][r] - mrow[r]);
          rsum[r] += p;
          Ps[w][(g * 8 + r) * 64 + nt * 16 + nl] = (_Float16)p;
        }
#pragma unroll
      for (int r = 0; r < 8; ++r) {
        float v = rsum[r];
        v += __shfl_xor(v, 1, 32);
        v += __shfl_xor(v, 2, 32);
        v += __shfl_xor(v, 4, 32);
        v += __shfl_xor(v, 8, 32);
        lrow[r] += v;
      }
      // O += P V   (per-wave LDS re-layout; DS ops in-order per wave)
      const v16h pf0 = load_a_frag(&Ps[w][0], 64, 0, lane);
      const v16h pf1 = load_a_frag(&Ps[w][0], 64, 32, lane);
#pragma unroll
      for (int nt = 0; nt < 4; ++nt) {
        const v16h v0 = load_b_frag(Vts + nt * 16 * 64, 64, 0, lane);
        O[nt] = wmma_f16(pf0, v0, O[nt]);
        const v16h v1 = load_b_frag(Vts + nt * 16 * 64, 64, 32, lane);
        O[nt] = wmma_f16(pf1, v1, O[nt]);
      }
    }
    __syncthreads();                                // tile consumed: reusable
  }

#pragma unroll
  for (int nt = 0; nt < 4; ++nt)
#pragma unroll
    for (int r = 0; r < 8; ++r) {
      const size_t row = (size_t)b * 2048 + row0 + w * 16 + g * 8 + r;
      out[row * 64 + nt * 16 + nl] = O[nt][r] / lrow[r];
    }
}

extern "C" void kernel_launch(void* const* d_in, const int* in_sizes, int n_in,
                              void* d_out, int out_size, void* d_ws,
                              size_t ws_size, hipStream_t stream) {
  const float* x  = (const float*)d_in[0];
  const float* Wk = (const float*)d_in[1];
  const float* Wq = (const float*)d_in[2];
  const float* Wv = (const float*)d_in[3];
  float* out = (float*)d_out;

  const size_t BT = (size_t)32 * 2048;
  _Float16* Qw = (_Float16*)d_ws;
  _Float16* Kw = Qw + BT * 64;
  _Float16* Vw = Kw + BT * 64;   // transposed layout [B][64][2048]

  proj_qkv_kernel<<<512, 256, 0, stream>>>(x, Wk, Wq, Wv, Qw, Kw, Vw);
  attn_kernel<<<512, 256, 0, stream>>>(Qw, Kw, Vw, out);
}
